// NeRF_85736137163457
// MI455X (gfx1250) — compile-verified, matrix operand order checked by reading, not run on hardware
//
#include <hip/hip_runtime.h>
#include <hip/hip_bf16.h>

typedef __attribute__((ext_vector_type(16))) _Float16 v16h;
typedef __attribute__((ext_vector_type(8)))  float    v8f;

// One wave32 handles 16 rays. Loop over S samples:
//   layer1 via affine trick (o@W1 + t*(d@W1), VALU FMAs),
//   layer2 via 2x v_wmma_f32_16x16x32_f16 (M=16 rays, K=64 hidden, N=4 live cols),
//   alpha compositing in registers using the WMMA D-matrix layout.
// The A-matrix pack is software-pipelined one sample ahead so the
// WMMA->VALU hazard shadow is filled with independent pack work.
__global__ __launch_bounds__(256) void nerf_wmma_kernel(
    const float* __restrict__ origins,     // [B,3]
    const float* __restrict__ directions,  // [B,3]
    const float* __restrict__ nearp,       // [1]
    const float* __restrict__ farp,        // [1]
    const float* __restrict__ W1,          // [3,64]
    const float* __restrict__ b1,          // [64]
    const float* __restrict__ W2,          // [64,4]
    const float* __restrict__ b2,          // [4]
    const int*   __restrict__ nsamp,       // [1]
    float*       __restrict__ out,         // [B,3]
    int B)
{
    const int lane    = threadIdx.x & 31;
    const int wave    = blockIdx.x * (blockDim.x >> 5) + (threadIdx.x >> 5);
    const int rayBase = wave * 16;
    if (rayBase >= B) return;              // wave-uniform exit (EXEC stays all-1s)

    const int   c     = lane & 15;         // A-side: ray-in-tile; B/D-side: output column
    const int   hi    = (lane >> 4) & 1;   // upper half-wave flag
    const float nearv = *nearp;
    const float farv  = *farp;
    const int   S     = *nsamp;
    const float delta = (farv - nearv) / (float)S;

    // ---- per-ray normalized direction & affine layer-1 precompute -------------
    int ray = rayBase + c; if (ray >= B) ray = B - 1;
    const float ox = origins[ray*3+0], oy = origins[ray*3+1], oz = origins[ray*3+2];
    float dx = directions[ray*3+0], dy = directions[ray*3+1], dz = directions[ray*3+2];
    const float inv = 1.0f / fmaxf(sqrtf(dx*dx + dy*dy + dz*dz), 1e-12f);
    dx *= inv; dy *= inv; dz *= inv;

    // ob/db hold (o@W1 + b1) and (d@W1) at hidden index 16t + r + 8*hi,
    // for this lane's ray (c). This is exactly the f16 A-matrix 16x32 layout.
    float ob[4][8], db[4][8];
    #pragma unroll
    for (int t = 0; t < 4; ++t) {
        #pragma unroll
        for (int r = 0; r < 8; ++r) {
            const int h = 16*t + r + 8*hi;
            const float w0 = W1[h], w1 = W1[64 + h], w2 = W1[128 + h];
            ob[t][r] = fmaf(ox, w0, fmaf(oy, w1, fmaf(oz, w2, b1[h])));
            db[t][r] = fmaf(dx, w0, fmaf(dy, w1, dz * w2));
        }
    }

    // ---- W2 (64x4, padded to Nx16) in f16 B-matrix layout, two K=32 chunks ----
    const float colMask = (c < 4) ? 1.0f : 0.0f;   // branchless: keeps EXEC full
    const int   cc      = c & 3;
    v16h B0, B1;
    #pragma unroll
    for (int e = 0; e < 16; ++e) {
        const int k0 = (e < 8) ? (8*hi + e) : (16 + 8*hi + (e - 8));
        B0[e] = (_Float16)(W2[(k0     ) * 4 + cc] * colMask);
        B1[e] = (_Float16)(W2[(k0 + 32) * 4 + cc] * colMask);
    }
    const float b2c = b2[cc] * colMask;

    // Per-lane compositing state for rays (r + 8*hi), this lane's channel c.
    float T[8], acc[8];
    #pragma unroll
    for (int r = 0; r < 8; ++r) { T[r] = 1.0f; acc[r] = 0.0f; }

    // layer-1 pack: h = relu(ob + m*db) into the WMMA f16 A layout
    auto packA = [&](float m, v16h& A0, v16h& A1) {
        #pragma unroll
        for (int r = 0; r < 8; ++r) {
            A0[r    ] = (_Float16)fmaxf(fmaf(m, db[0][r], ob[0][r]), 0.0f);
            A0[r + 8] = (_Float16)fmaxf(fmaf(m, db[1][r], ob[1][r]), 0.0f);
            A1[r    ] = (_Float16)fmaxf(fmaf(m, db[2][r], ob[2][r]), 0.0f);
            A1[r + 8] = (_Float16)fmaxf(fmaf(m, db[3][r], ob[3][r]), 0.0f);
        }
    };

    // ---- main loop over samples (A-pack pipelined one step ahead) -------------
    v16h A0c, A1c;
    packA(fmaf(delta, 0.5f, nearv), A0c, A1c);

    #pragma unroll 1
    for (int s = 0; s < S; ++s) {
        // layer 2: [16 rays x 64] @ [64 x 16] via two chained WMMAs (K=32 each)
        v8f O = {0.f,0.f,0.f,0.f,0.f,0.f,0.f,0.f};
        O = __builtin_amdgcn_wmma_f32_16x16x32_f16(false, A0c, false, B0,
                                                   (short)0, O, false, false);
        O = __builtin_amdgcn_wmma_f32_16x16x32_f16(false, A1c, false, B1,
                                                   (short)0, O, false, false);

        // Next iteration's A pack: independent of O -> fills the WMMA->VALU
        // hazard shadow and co-executes with the matrix pipe.
        v16h A0n, A1n;
        packA(fmaf(delta, (float)(s + 1) + 0.5f, nearv), A0n, A1n);

        // compositing: D layout => VGPR r holds ray (r + 8*hi), column c.
        #pragma unroll
        for (int r = 0; r < 8; ++r) {
            const float val = __expf(O[r] + b2c);              // color (c<3) / sigma (c==3)
            const float sa  = 1.0f - __expf(-val * delta);     // alpha (valid on c==3)
            const float a   = __shfl(sa, (lane & 16) | 3, 32); // broadcast sigma lane
            T[r] *= (1.0f - a);                                // inclusive cumprod
            acc[r] = fmaf(T[r] * a, val, acc[r]);              // w * color
        }

        A0c = A0n; A1c = A1n;
    }

    // ---- write composited RGB -------------------------------------------------
    if (c < 3) {
        #pragma unroll
        for (int r = 0; r < 8; ++r) {
            const int ridx = rayBase + r + 8*hi;
            if (ridx < B) out[ridx * 3 + c] = acc[r];
        }
    }
}

extern "C" void kernel_launch(void* const* d_in, const int* in_sizes, int n_in,
                              void* d_out, int out_size, void* d_ws, size_t ws_size,
                              hipStream_t stream) {
    const float* origins    = (const float*)d_in[0];
    const float* directions = (const float*)d_in[1];
    const float* nearp      = (const float*)d_in[2];
    const float* farp       = (const float*)d_in[3];
    const float* W1         = (const float*)d_in[4];
    const float* b1         = (const float*)d_in[5];
    const float* W2         = (const float*)d_in[6];
    const float* b2         = (const float*)d_in[7];
    const int*   nsamp      = (const int*)d_in[8];
    float*       out        = (float*)d_out;

    const int B      = in_sizes[0] / 3;          // rays
    const int waves  = (B + 15) / 16;            // 16 rays per wave32
    const int blocks = (waves + 7) / 8;          // 8 waves (256 threads) per block

    nerf_wmma_kernel<<<blocks, 256, 0, stream>>>(origins, directions, nearp, farp,
                                                 W1, b1, W2, b2, nsamp, out, B);
}
